// AKT_8684423873281
// MI455X (gfx1250) — matrix operand order repaired and checked by reading, not compiled, not run on hardware
//
#include <hip/hip_runtime.h>
#include <hip/hip_bf16.h>

// ---------------------------------------------------------------------------
// AKT transformer forward for MI455X (gfx1250, wave32, WMMA).
// All GEMMs run on v_wmma_f32_16x16x32_f16 (f16 in, f32 accumulate).
// ---------------------------------------------------------------------------

typedef _Float16 half16 __attribute__((ext_vector_type(16)));
typedef _Float16 half8  __attribute__((ext_vector_type(8)));
typedef float    f32x8  __attribute__((ext_vector_type(8)));

static constexpr int AB  = 32;        // batch
static constexpr int AS  = 512;       // sequence
static constexpr int AD  = 512;       // model dim
static constexpr int AH  = 8;         // heads
static constexpr int ADK = AD / AH;   // 64
static constexpr int AF  = 2048;      // ffn dim
static constexpr int AM  = AB * AS;   // 16384 rows
static constexpr int ABH = AB * AH;   // 256 batch-heads
static constexpr int HCHUNK = 32;     // heads per attention chunk
static constexpr int MCHUNK = 4096;   // rows per FFN chunk

__device__ __forceinline__ half16 frag_cat(half8 lo, half8 hi) {
    return __builtin_shufflevector(lo, hi, 0,1,2,3,4,5,6,7,8,9,10,11,12,13,14,15);
}

// ---------------------------------------------------------------------------
// WMMA GEMM: C[m][n] = scale * sum_k A[m][k] * Bt[n][k] + bias[n]
//   A  : f16 [M,K] row-major         (batched via sA)
//   Bt : f16 [N,K] row-major (B^T)   (batched via sB)
//   Cf : optional f32 [M,N]          (batched via sC)
//   Ch : optional f16 [M,N]          (batched via sC)
// Block = 128 threads = 4 waves (2x2).
// Wave tile = 64 x (16*NT)  ->  block tile = 128 x (32*NT).
//   NT=4 (64x64 wave tile, 16 WMMA / K-step) for N % 128 == 0 GEMMs,
//   NT=2 (64x32 wave tile,  8 WMMA / K-step) for the N=64 attn-V GEMM.
// Requires M%128==0, N%(32*NT)==0, K%32==0 (true for all shapes used here).
//
// Fragment layouts (CDNA5 ISA 7.12.2, 16-bit, wave32):
//   A 16x32: lane L holds row M=(L&15); lanes 0-15 carry K 0..7 & 16..23,
//            lanes 16-31 carry K 8..15 & 24..31  -> two b128 loads per lane.
//   B 32x16: lane L holds col N=(L&15); lanes 0-15 carry K 0..7 & 16..23,
//            lanes 16-31 carry K 8..15 & 24..31  -> two b128 loads per lane.
//   C/D 16x16 f32: elem e of v8f -> row e + (L>=16 ? 8:0), col (L&15).
// ---------------------------------------------------------------------------
template <int NT, bool RELU>
__global__ __launch_bounds__(128) void wmma_gemm(
    const _Float16* __restrict__ Ab, const _Float16* __restrict__ Btb,
    const float* __restrict__ bias,
    float* __restrict__ Cf, _Float16* __restrict__ Ch,
    int Mdim, int Ndim, int Kdim, float scale,
    long long sA, long long sB, long long sC)
{
    const int lane   = threadIdx.x & 31;
    const int wave   = threadIdx.x >> 5;
    const int laneHi = lane >> 4;
    const int r      = lane & 15;

    const _Float16* A  = Ab  + (size_t)blockIdx.z * (size_t)sA;
    const _Float16* Bt = Btb + (size_t)blockIdx.z * (size_t)sB;

    const int m0 = blockIdx.y * 128       + (wave >> 1) * 64;
    const int n0 = blockIdx.x * (32 * NT) + (wave & 1)  * (16 * NT);

    const _Float16* aRow[4];
    const _Float16* bRow[NT];
#pragma unroll
    for (int i = 0; i < 4; ++i)
        aRow[i] = A + (size_t)(m0 + i * 16 + r) * Kdim + laneHi * 8;
#pragma unroll
    for (int j = 0; j < NT; ++j)
        bRow[j] = Bt + (size_t)(n0 + j * 16 + r) * Kdim + laneHi * 8;

    f32x8 zero = {};
    f32x8 acc[4][NT];
#pragma unroll
    for (int i = 0; i < 4; ++i)
#pragma unroll
        for (int j = 0; j < NT; ++j) acc[i][j] = zero;

    for (int kk = 0; kk < Kdim; kk += 32) {
        // Unconditional speculative prefetch of the next K panel; invalid
        // translations are silently dropped (ISA 10.5), keeps loop branch-free.
        __builtin_prefetch(aRow[0] + kk + 64, 0, 1);
        __builtin_prefetch(bRow[0] + kk + 64, 0, 1);

        half16 af[4], bf[NT];
#pragma unroll
        for (int i = 0; i < 4; ++i) {
            const _Float16* p = aRow[i] + kk;
            af[i] = frag_cat(*(const half8*)p, *(const half8*)(p + 16));
        }
#pragma unroll
        for (int j = 0; j < NT; ++j) {
            const _Float16* p = bRow[j] + kk;
            bf[j] = frag_cat(*(const half8*)p, *(const half8*)(p + 16));
        }
#pragma unroll
        for (int i = 0; i < 4; ++i)
#pragma unroll
            for (int j = 0; j < NT; ++j)
                acc[i][j] = __builtin_amdgcn_wmma_f32_16x16x32_f16(
                    false, af[i], false, bf[j], (short)0, acc[i][j], false, false);
    }

    float*    Cfb = Cf ? Cf + (size_t)blockIdx.z * (size_t)sC : nullptr;
    _Float16* Chb = Ch ? Ch + (size_t)blockIdx.z * (size_t)sC : nullptr;
#pragma unroll
    for (int i = 0; i < 4; ++i) {
#pragma unroll
        for (int j = 0; j < NT; ++j) {
            const int col  = n0 + j * 16 + r;
            const int rowb = m0 + i * 16 + laneHi * 8;
            const float bv = bias ? bias[col] : 0.f;
#pragma unroll
            for (int e = 0; e < 8; ++e) {
                float v = acc[i][j][e] * scale + bv;
                if (RELU) v = fmaxf(v, 0.f);
                const size_t off = (size_t)(rowb + e) * Ndim + col;
                if (Cfb) Cfb[off] = v;
                if (Chb) Chb[off] = (_Float16)v;
            }
        }
    }
}

// ---------------------------------------------------------------------------
// Pointwise / reshape helpers
// ---------------------------------------------------------------------------
__global__ __launch_bounds__(256) void k_transpose_f16(
    const float* __restrict__ W, _Float16* __restrict__ Wt, int K, int N)
{   // W [K,N] f32 -> Wt [N,K] f16
    const int i = blockIdx.x * 256 + threadIdx.x;
    const int kk = i / N, nn = i % N;
    Wt[(size_t)nn * K + kk] = (_Float16)W[i];
}

__global__ __launch_bounds__(256) void k_convert_f16(
    const float* __restrict__ in, _Float16* __restrict__ out)
{
    const int i = blockIdx.x * 256 + threadIdx.x;
    out[i] = (_Float16)in[i];
}

__global__ __launch_bounds__(256) void k_split_q(
    const float* __restrict__ proj, _Float16* __restrict__ qh)
{   // [B,S,D] f32 -> [B,H,S,dk] f16
    const int i = blockIdx.x * 256 + threadIdx.x;
    const int b = i / (AS * AD), rr = i % (AS * AD);
    const int s = rr / AD, c = rr % AD;
    const int h = c / ADK, d = c % ADK;
    qh[(((size_t)b * AH + h) * AS + s) * ADK + d] = (_Float16)proj[i];
}

__global__ __launch_bounds__(256) void k_split_vT(
    const float* __restrict__ proj, _Float16* __restrict__ vt)
{   // [B,S,D] f32 -> [B,H,dk,S] f16 (V transposed per head, feeds WMMA B^T)
    const int i = blockIdx.x * 256 + threadIdx.x;
    const int b = i / (AS * AD), rr = i % (AS * AD);
    const int s = rr / AD, c = rr % AD;
    const int h = c / ADK, d = c % ADK;
    vt[(((size_t)b * AH + h) * ADK + d) * AS + s] = (_Float16)proj[i];
}

__global__ __launch_bounds__(256) void k_merge(
    const float* __restrict__ ctx, _Float16* __restrict__ merged)
{   // [B,H,S,dk] f32 -> [B,S,D] f16
    const int i = blockIdx.x * 256 + threadIdx.x;
    const int bh = i / (AS * ADK), rr = i % (AS * ADK);
    const int s = rr / ADK, d = rr % ADK;
    const int b = bh / AH, h = bh % AH;
    merged[((size_t)b * AS + s) * AD + h * ADK + d] = (_Float16)ctx[i];
}

// Residual + LayerNorm over D=512; writes f32 (+optional f16) outputs.
__global__ __launch_bounds__(256) void k_add_ln(
    const float* __restrict__ res, const float* __restrict__ add,
    const float* __restrict__ g, const float* __restrict__ b,
    float* __restrict__ out32, _Float16* __restrict__ out16)
{
    const int row = blockIdx.x, t = threadIdx.x;
    const float* r0 = res + (size_t)row * AD;
    const float* a0 = add + (size_t)row * AD;
    const float x0 = r0[t] + a0[t];
    const float x1 = r0[t + 256] + a0[t + 256];
    __shared__ float red[256];
    red[t] = x0 + x1; __syncthreads();
    for (int off = 128; off > 0; off >>= 1) { if (t < off) red[t] += red[t + off]; __syncthreads(); }
    const float mean = red[0] * (1.f / AD); __syncthreads();
    const float d0 = x0 - mean, d1 = x1 - mean;
    red[t] = d0 * d0 + d1 * d1; __syncthreads();
    for (int off = 128; off > 0; off >>= 1) { if (t < off) red[t] += red[t + off]; __syncthreads(); }
    const float rstd = rsqrtf(red[0] * (1.f / AD) + 1e-5f);
    const float y0 = d0 * rstd * g[t]       + b[t];
    const float y1 = d1 * rstd * g[t + 256] + b[t + 256];
    float* o = out32 + (size_t)row * AD;
    o[t] = y0; o[t + 256] = y1;
    if (out16) {
        _Float16* oh = out16 + (size_t)row * AD;
        oh[t] = (_Float16)y0; oh[t + 256] = (_Float16)y1;
    }
}

// ---------------------------------------------------------------------------
// AKT monotonic attention row kernel. One 512-thread block per (head, q) row.
//   p     = softmax(mask ? s : -1e32) * mask       (masked-max softmax)
//   dist  = sqrt(clip((total - cumsum(p)) * |q-k|))
//   te    = clip(exp(-softplus(gamma) * dist), 1e-5, 1e5)
//   attn  = softmax(mask ? s*te : -1e32) ; row 0 zeroed when mask_k==0
// Cumsum via Hillis-Steele scan in LDS; reductions via LDS trees.
// ---------------------------------------------------------------------------
__global__ __launch_bounds__(512) void k_attn_rows(
    const float* __restrict__ scores, _Float16* __restrict__ attn,
    const float* __restrict__ gam, int bh0, int mask_k)
{
    const int k = threadIdx.x;
    const int q = blockIdx.x;
    const int bhl = blockIdx.y;
    const int h = (bh0 + bhl) % AH;
    const float s = scores[((size_t)bhl * AS + q) * AS + k];
    const bool msk = (k <= q + mask_k - 1);

    __shared__ float sa[512], sb[512];

    // masked max
    sa[k] = msk ? s : -1e32f; __syncthreads();
    for (int off = 256; off > 0; off >>= 1) {
        if (k < off) sa[k] = fmaxf(sa[k], sa[k + off]);
        __syncthreads();
    }
    const float mx = sa[0]; __syncthreads();

    // unnormalized masked exp; inclusive scan gives cumsum and total
    const float e = msk ? __expf(s - mx) : 0.f;
    sa[k] = e; __syncthreads();
    float* src = sa; float* dst = sb;
    for (int off = 1; off < 512; off <<= 1) {
        float v = src[k];
        if (k >= off) v += src[k - off];
        dst[k] = v; __syncthreads();
        float* tmp = src; src = dst; dst = tmp;
    }
    const float cum = src[k];
    const float tot = src[AS - 1];
    __syncthreads();

    const float denom    = fmaxf(tot, 1e-30f);
    const float distcum  = cum / denom;
    const float disttot  = tot / denom;
    const float pos      = fabsf((float)(q - k));
    const float dist     = sqrtf(fmaxf((disttot - distcum) * pos, 0.f));
    const float gm       = gam[h];
    const float sp       = (gm > 20.f) ? gm : log1pf(__expf(gm));   // softplus
    float te = __expf(dist * (-sp));
    te = fminf(fmaxf(te, 1e-5f), 1e5f);
    const float s2 = msk ? s * te : -1e32f;

    // second softmax
    sa[k] = s2; __syncthreads();
    for (int off = 256; off > 0; off >>= 1) {
        if (k < off) sa[k] = fmaxf(sa[k], sa[k + off]);
        __syncthreads();
    }
    const float mx2 = sa[0]; __syncthreads();
    const float e2 = __expf(s2 - mx2);   // all-masked row: exp(0)=1 each
    sa[k] = e2; __syncthreads();
    for (int off = 256; off > 0; off >>= 1) {
        if (k < off) sa[k] += sa[k + off];
        __syncthreads();
    }
    float a = e2 / sa[0];
    if (mask_k == 0 && q == 0) a = 0.f;   // zero_pad row
    attn[((size_t)bhl * AS + q) * AS + k] = (_Float16)a;
}

// ---------------------------------------------------------------------------
// Host orchestration
// ---------------------------------------------------------------------------
struct LayerP {
    const float *bk, *bv, *bo, *gam, *l1g, *l1b, *bf1, *bf2, *l2g, *l2b;
    _Float16 *WkT, *WvT, *WoT, *Wf1T, *Wf2T;
};

extern "C" void kernel_launch(void* const* d_in, const int* in_sizes, int n_in,
                              void* d_out, int out_size, void* d_ws, size_t ws_size,
                              hipStream_t stream) {
    (void)n_in; (void)out_size; (void)ws_size;

    // ---- workspace bump allocator (256B aligned) ----
    char* cur = (char*)d_ws;
    auto alloc = [&](size_t n) -> char* {
        char* r = cur; cur += (n + 255) & ~(size_t)255; return r;
    };

    // ---- scratch buffers (reused across layers) ----
    float*     projqk = (float*)alloc((size_t)AM * AD * 4);
    float*     projv  = (float*)alloc((size_t)AM * AD * 4);
    _Float16*  Qh     = (_Float16*)alloc((size_t)AM * AD * 2);
    _Float16*  Vt     = (_Float16*)alloc((size_t)AM * AD * 2);
    float*     ctx    = (float*)alloc((size_t)AM * AD * 4);
    _Float16*  merged = (_Float16*)alloc((size_t)AM * AD * 2);
    float*     att32  = (float*)alloc((size_t)AM * AD * 4);
    float*     x1_32  = (float*)alloc((size_t)AM * AD * 4);
    _Float16*  x1_16  = (_Float16*)alloc((size_t)AM * AD * 2);
    float*     scoresC = (float*)alloc((size_t)HCHUNK * AS * AS * 4);
    _Float16*  attnC   = (_Float16*)alloc((size_t)HCHUNK * AS * AS * 2);
    _Float16*  midh    = (_Float16*)alloc((size_t)MCHUNK * AF * 2);
    float*     y32    = (float*)alloc((size_t)AM * AD * 4);
    _Float16*  y16    = (_Float16*)alloc((size_t)AM * AD * 2);
    float*     x32    = (float*)alloc((size_t)AM * AD * 4);
    _Float16*  x16    = (_Float16*)alloc((size_t)AM * AD * 2);
    float*     fbuf   = projqk;   // reuse: projqk dead after split, fbuf live in FFN

    // ---- GEMM launch helper: pick widest wave tile the N dim allows ----
    auto gemm = [&](const _Float16* A, const _Float16* Bt, const float* bias,
                    float* Cf, _Float16* Ch, int Mg, int Ng, int Kg, float scale,
                    long long sA, long long sB, long long sC, int batch, bool relu) {
        dim3 block(128);
        if (Ng % 128 == 0) {
            dim3 grid(Ng / 128, Mg / 128, batch);
            if (relu)
                wmma_gemm<4, true><<<grid, block, 0, stream>>>(A, Bt, bias, Cf, Ch,
                                                               Mg, Ng, Kg, scale, sA, sB, sC);
            else
                wmma_gemm<4, false><<<grid, block, 0, stream>>>(A, Bt, bias, Cf, Ch,
                                                                Mg, Ng, Kg, scale, sA, sB, sC);
        } else {
            dim3 grid(Ng / 64, Mg / 128, batch);
            wmma_gemm<2, false><<<grid, block, 0, stream>>>(A, Bt, bias, Cf, Ch,
                                                            Mg, Ng, Kg, scale, sA, sB, sC);
        }
    };

    // ---- weight prep: transpose to f16 [N,K]; weights stay L2-resident ----
    auto make_layer = [&](int ib, int i) -> LayerP {
        LayerP L;
        const float* Wk  = (const float*)d_in[ib + 0]  + (size_t)i * AD * AD;
        L.bk   = (const float*)d_in[ib + 1]  + (size_t)i * AD;
        const float* Wv  = (const float*)d_in[ib + 2]  + (size_t)i * AD * AD;
        L.bv   = (const float*)d_in[ib + 3]  + (size_t)i * AD;
        const float* Wo  = (const float*)d_in[ib + 4]  + (size_t)i * AD * AD;
        L.bo   = (const float*)d_in[ib + 5]  + (size_t)i * AD;
        L.gam  = (const float*)d_in[ib + 6]  + (size_t)i * AH;
        L.l1g  = (const float*)d_in[ib + 7]  + (size_t)i * AD;
        L.l1b  = (const float*)d_in[ib + 8]  + (size_t)i * AD;
        const float* Wf1 = (const float*)d_in[ib + 9]  + (size_t)i * AD * AF;
        L.bf1  = (const float*)d_in[ib + 10] + (size_t)i * AF;
        const float* Wf2 = (const float*)d_in[ib + 11] + (size_t)i * AF * AD;
        L.bf2  = (const float*)d_in[ib + 12] + (size_t)i * AD;
        L.l2g  = (const float*)d_in[ib + 13] + (size_t)i * AD;
        L.l2b  = (const float*)d_in[ib + 14] + (size_t)i * AD;
        L.WkT  = (_Float16*)alloc((size_t)AD * AD * 2);
        L.WvT  = (_Float16*)alloc((size_t)AD * AD * 2);
        L.WoT  = (_Float16*)alloc((size_t)AD * AD * 2);
        L.Wf1T = (_Float16*)alloc((size_t)AD * AF * 2);
        L.Wf2T = (_Float16*)alloc((size_t)AF * AD * 2);
        k_transpose_f16<<<AD * AD / 256, 256, 0, stream>>>(Wk,  L.WkT,  AD, AD);
        k_transpose_f16<<<AD * AD / 256, 256, 0, stream>>>(Wv,  L.WvT,  AD, AD);
        k_transpose_f16<<<AD * AD / 256, 256, 0, stream>>>(Wo,  L.WoT,  AD, AD);
        k_transpose_f16<<<AD * AF / 256, 256, 0, stream>>>(Wf1, L.Wf1T, AD, AF);
        k_transpose_f16<<<AF * AD / 256, 256, 0, stream>>>(Wf2, L.Wf2T, AF, AD);
        return L;
    };

    const int n1 = in_sizes[2]  / (AD * AD);
    const int n2 = in_sizes[17] / (AD * AD);
    LayerP L1[8], L2[8];
    for (int i = 0; i < n1 && i < 8; ++i) L1[i] = make_layer(2, i);
    for (int i = 0; i < n2 && i < 8; ++i) L2[i] = make_layer(17, i);

    // ---- one transformer layer (q_in == k_in in every call of this model) ----
    auto run_layer = [&](int mask_k, bool apply_pos,
                         const float* q32, const _Float16* q16, const _Float16* v16,
                         const LayerP& L, float* out32, _Float16* out16) {
        // shared K/Q projection (kq_same) and V projection
        gemm(q16, L.WkT, L.bk, projqk, nullptr, AM, AD, AD, 1.f, 0, 0, 0, 1, false);
        gemm(v16, L.WvT, L.bv, projv,  nullptr, AM, AD, AD, 1.f, 0, 0, 0, 1, false);
        k_split_q <<<AM * AD / 256, 256, 0, stream>>>(projqk, Qh);
        k_split_vT<<<AM * AD / 256, 256, 0, stream>>>(projv,  Vt);

        // attention, HCHUNK heads at a time (bounds the S x S buffers)
        for (int c = 0; c < ABH; c += HCHUNK) {
            const _Float16* Qc = Qh + (size_t)c * AS * ADK;
            gemm(Qc, Qc, nullptr, scoresC, nullptr, AS, AS, ADK, 0.125f,
                 (long long)AS * ADK, (long long)AS * ADK, (long long)AS * AS,
                 HCHUNK, false);
            k_attn_rows<<<dim3(AS, HCHUNK), 512, 0, stream>>>(scoresC, attnC,
                                                              L.gam, c, mask_k);
            gemm(attnC, Vt + (size_t)c * ADK * AS, nullptr,
                 ctx + (size_t)c * AS * ADK, nullptr, AS, ADK, AS, 1.f,
                 (long long)AS * AS, (long long)ADK * AS, (long long)AS * ADK,
                 HCHUNK, false);
        }
        k_merge<<<AM * AD / 256, 256, 0, stream>>>(ctx, merged);
        gemm(merged, L.WoT, L.bo, att32, nullptr, AM, AD, AD, 1.f, 0, 0, 0, 1, false);

        if (apply_pos) {
            k_add_ln<<<AM, 256, 0, stream>>>(q32, att32, L.l1g, L.l1b, x1_32, x1_16);
            for (int mc = 0; mc < AM; mc += MCHUNK) {
                gemm(x1_16 + (size_t)mc * AD, L.Wf1T, L.bf1, nullptr, midh,
                     MCHUNK, AF, AD, 1.f, 0, 0, 0, 1, true);   // fused ReLU -> f16
                gemm(midh, L.Wf2T, L.bf2, fbuf + (size_t)mc * AD, nullptr,
                     MCHUNK, AD, AF, 1.f, 0, 0, 0, 1, false);
            }
            k_add_ln<<<AM, 256, 0, stream>>>(x1_32, fbuf, L.l2g, L.l2b, out32, out16);
        } else {
            k_add_ln<<<AM, 256, 0, stream>>>(q32, att32, L.l1g, L.l1b, out32, out16);
        }
    };

    // ---- forward ----
    const float* qa32 = (const float*)d_in[1];
    const float* qe32 = (const float*)d_in[0];
    k_convert_f16<<<AM * AD / 256, 256, 0, stream>>>(qa32, y16);
    k_convert_f16<<<AM * AD / 256, 256, 0, stream>>>(qe32, x16);

    // blocks_1: self-attention on y, mask=1, apply_pos=true
    const float* curY32 = qa32;
    for (int i = 0; i < n1 && i < 8; ++i) {
        run_layer(1, true, curY32, y16, y16, L1[i], y32, y16);
        curY32 = y32;
    }

    // blocks_2: alternating (1,x,x,x,no-pos) / (0,x,x,y,pos)
    const float* curX32 = qe32;
    bool flag_first = true;
    for (int i = 0; i < n2 && i < 8; ++i) {
        const bool last = (i == n2 - 1);
        float* o32 = last ? (float*)d_out : x32;
        if (flag_first) {
            run_layer(1, false, curX32, x16, x16, L2[i], o32, x16);
            flag_first = false;
        } else {
            run_layer(0, true, curX32, x16, y16, L2[i], o32, x16);
            flag_first = true;
        }
        curX32 = o32;
    }
}
